// out_heads_72988674228878
// MI455X (gfx1250) — matrix operand order repaired and checked by reading, not compiled, not run on hardware
//
#include <hip/hip_runtime.h>
#include <hip/hip_bf16.h>

// ---------------------------------------------------------------------------
// Routed 2-layer MLP (grouped GEMM), MI455X / gfx1250, wave32 + WMMA bf16.
//   y[i] = relu(x[i] @ W1[gid[i]] + b1[gid[i]]) @ W2[gid[i]] + b2[gid[i]]
// B=4096, G=8, IN=512, H=1024, OUT=64
// Only the routed expert is computed (1/8 of the reference FLOPs).
// ---------------------------------------------------------------------------

typedef __attribute__((ext_vector_type(16))) __bf16   v16bf;
typedef __attribute__((ext_vector_type(8)))  float    v8f;
typedef __attribute__((ext_vector_type(4)))  float    v4f;
typedef __attribute__((ext_vector_type(4)))  unsigned v4u;
typedef __attribute__((ext_vector_type(2)))  unsigned v2u;

#define B_N    4096
#define G_N    8
#define IN_N   512
#define H_N    1024
#define OUT_N  64
#define TILE_M 16
#define MAX_TILES (B_N / TILE_M + G_N)   // 264 (worst-case padded tiles)

// ---- workspace layout (int32 units) ---------------------------------------
#define WS_NUMTILES    0
#define WS_COUNTS      1                   // [8]
#define WS_CURSOR      9                   // [8]
#define WS_TILE_EXPERT 32                  // [MAX_TILES]
#define WS_TILE_ROW0   320                 // [MAX_TILES]
#define WS_BUCKET      600                 // [8]
#define WS_PERM        640                 // [B_N + 16*G_N]
#define WS_TOTAL       (WS_PERM + B_N + TILE_M * G_N)   // 4864 ints = 19 KB

// ---------------------------------------------------------------------------
// Routing kernels
// ---------------------------------------------------------------------------
__global__ void route_init(int* ws) {
    int i = blockIdx.x * blockDim.x + threadIdx.x;
    if (i < WS_TOTAL) ws[i] = (i >= WS_PERM) ? -1 : 0;
}

__global__ void route_hist(const int* __restrict__ gid, int* ws) {
    int i = blockIdx.x * blockDim.x + threadIdx.x;
    if (i < B_N) atomicAdd(&ws[WS_COUNTS + gid[i]], 1);
}

__global__ void route_plan(int* ws) {
    if (blockIdx.x == 0 && threadIdx.x == 0) {
        int base = 0, T = 0;
        for (int g = 0; g < G_N; ++g) {
            ws[WS_BUCKET + g] = base;
            int c  = ws[WS_COUNTS + g];
            int nt = (c + TILE_M - 1) / TILE_M;
            for (int t = 0; t < nt; ++t) {
                ws[WS_TILE_EXPERT + T] = g;
                ws[WS_TILE_ROW0   + T] = base + t * TILE_M;
                ++T;
            }
            base += nt * TILE_M;
        }
        ws[WS_NUMTILES] = T;
    }
}

__global__ void route_scatter(const int* __restrict__ gid, int* ws) {
    int i = blockIdx.x * blockDim.x + threadIdx.x;
    if (i < B_N) {
        int g = gid[i];
        int p = atomicAdd(&ws[WS_CURSOR + g], 1);
        ws[WS_PERM + ws[WS_BUCKET + g] + p] = i;
    }
}

// ---------------------------------------------------------------------------
// Fused MoE MLP kernel: one workgroup = one 16-row tile of one expert.
// 256 threads = 8 wave32. bf16 WMMA (16x16x32), f32 accumulation.
// ---------------------------------------------------------------------------
union Frag  { v4u q[2]; unsigned u[8]; v16bf v; };
union Pack4 { __bf16 h[4]; v2u u2; };

static __device__ inline unsigned pack2_bf16(float lo, float hi) {
    union { __bf16 h[2]; unsigned u; } p;
    p.h[0] = (__bf16)lo;
    p.h[1] = (__bf16)hi;
    return p.u;
}

__launch_bounds__(256, 1)
__global__ void moe_mlp_kernel(const int*   __restrict__ ws,
                               const float* __restrict__ x,
                               const float* __restrict__ W1,
                               const float* __restrict__ b1,
                               const float* __restrict__ W2,
                               const float* __restrict__ b2,
                               float*       __restrict__ out)
{
    const int t = blockIdx.x;
    if (t >= ws[WS_NUMTILES]) return;
    const int g    = ws[WS_TILE_EXPERT + t];
    const int row0 = ws[WS_TILE_ROW0   + t];

    const int  tid  = threadIdx.x;
    const int  lane = tid & 31;
    const int  wv   = tid >> 5;          // wave id 0..7
    const int  mcol = lane & 15;         // M for A-frag, N for B-frag / C/D
    const bool hi   = lane >= 16;
    const int  kbA  = hi ? 16 : 0;       // 16-bit A frag K base
    const int  kbB  = hi ? 16 : 0;       // 16-bit B frag K base

    // ---- LDS (strides chosen for 16B-aligned, bank-conflict-free b128) ----
    constexpr int XS  = IN_N + 8;        // 520 bf16/row: 1040B (16B align), bank +4/lane
    constexpr int WTS = 40;              // W1-slab^T: 80B/row (16B align), n*20u mod 64 sweeps banks
    constexpr int HS  = H_N + 8;         // 1032 bf16/row: 2064B (16B align), bank +4/lane
    __shared__ __bf16 s_x [TILE_M * XS];     // x tile    [16][IN]   bf16 (16.3 KB)
    __shared__ __bf16 s_wT[H_N * WTS];       // W1 slab^T [1024][32] bf16 (80 KB)
    __shared__ __bf16 s_h [TILE_M * HS];     // h tile    [16][H]    bf16 (32.3 KB)
    __shared__ int    s_rows[TILE_M];

    if (tid < TILE_M) s_rows[tid] = ws[WS_PERM + row0 + tid];
    __syncthreads();

    // ---- stage x tile as bf16 via float4 loads (padding rows -> 0) ----
    for (int i = tid * 4; i < TILE_M * IN_N; i += 256 * 4) {
        int r = i >> 9;                  // / IN_N
        int k = i & (IN_N - 1);          // multiple of 4
        int row = s_rows[r];
        v4f v = {0.0f, 0.0f, 0.0f, 0.0f};
        if (row >= 0) v = *(const v4f*)(x + (size_t)row * IN_N + k);
        Pack4 p;
        p.h[0] = (__bf16)v.x; p.h[1] = (__bf16)v.y;
        p.h[2] = (__bf16)v.z; p.h[3] = (__bf16)v.w;
        *(v2u*)(s_x + r * XS + k) = p.u2;     // 8B-aligned b64 store
    }

    const float* W1g = W1 + (size_t)g * IN_N * H_N;

    // per-wave: 8 output tiles of 16x16 covering 128 columns of H
    v8f acc[8] = {};

    const unsigned* sxu = (const unsigned*)s_x;
    const unsigned* swu = (const unsigned*)s_wT;

    // =================== layer 1: [16,512] x [512,1024] ===================
    for (int k0 = 0; k0 < IN_N; k0 += 32) {
        __syncthreads();                 // protect s_wT WAR (also covers s_x stage)
        // stage W1[k0:k0+32, :] transposed -> s_wT[n][kk]; float4 global reads
        for (int i = tid * 4; i < 32 * H_N; i += 256 * 4) {
            int kk = i >> 10;            // / H_N
            int n  = i & (H_N - 1);      // multiple of 4
            v4f w = *(const v4f*)(W1g + (size_t)(k0 + kk) * H_N + n);
            s_wT[(n + 0) * WTS + kk] = (__bf16)w.x;
            s_wT[(n + 1) * WTS + kk] = (__bf16)w.y;
            s_wT[(n + 2) * WTS + kk] = (__bf16)w.z;
            s_wT[(n + 3) * WTS + kk] = (__bf16)w.w;
        }
        __syncthreads();

        // A fragment (shared by all 8 N-tiles of this wave): 2x ds_load_b128
        Frag a;
        {
            const v4u* pa = (const v4u*)(sxu + ((mcol * XS + k0 + kbA) >> 1));
            a.q[0] = pa[0];              // K = kbA+{0..7}
            a.q[1] = pa[1];              // K = kbA+{8..15}
        }
        // Batch all 8 B fragments (distinct regs -> overlapped ds latency)
        Frag b[8];
        #pragma unroll
        for (int j = 0; j < 8; ++j) {
            int n = wv * 128 + j * 16 + mcol;
            const v4u* pb = (const v4u*)(swu + n * (WTS / 2) + (kbB >> 1));
            b[j].q[0] = pb[0];
            b[j].q[1] = pb[1];
        }
        #pragma unroll
        for (int j = 0; j < 8; ++j)
            acc[j] = __builtin_amdgcn_wmma_f32_16x16x32_bf16(
                         false, a.v, false, b[j].v, (short)0, acc[j], false, false);
    }

    // ---- epilogue 1: bias + ReLU (f32), store h tile as bf16 ----
    const float* b1g = b1 + (size_t)g * H_N;
    #pragma unroll
    for (int j = 0; j < 8; ++j) {
        int n = wv * 128 + j * 16 + mcol;
        float bias = b1g[n];
        #pragma unroll
        for (int r = 0; r < 8; ++r) {
            float v = acc[j][r] + bias;
            v = v > 0.0f ? v : 0.0f;
            int row = r + (hi ? 8 : 0);  // C/D layout: vgpr r -> M=r (+8 for hi lanes)
            s_h[row * HS + n] = (__bf16)v;
        }
    }
    __syncthreads();

    // =================== layer 2: [16,1024] x [1024,64] ===================
    if (wv < 4) {
        const float* W2g = W2 + (size_t)g * H_N * OUT_N;
        const unsigned* shu = (const unsigned*)s_h;
        const int nn = wv * 16 + mcol;   // output column 0..63
        v8f acc2 = {};
        for (int k0 = 0; k0 < H_N; k0 += 32) {
            Frag a;
            {
                const v4u* pa = (const v4u*)(shu + ((mcol * HS + k0 + kbA) >> 1));
                a.q[0] = pa[0];
                a.q[1] = pa[1];
            }
            Frag b;                       // W2 column nn, K pairs (L2-resident)
            #pragma unroll
            for (int i = 0; i < 8; ++i) {
                int k = k0 + kbB + 2 * i;
                b.u[i] = pack2_bf16(W2g[(size_t)k * OUT_N + nn],
                                    W2g[(size_t)(k + 1) * OUT_N + nn]);
            }
            acc2 = __builtin_amdgcn_wmma_f32_16x16x32_bf16(
                       false, a.v, false, b.v, (short)0, acc2, false, false);
        }
        float bias = b2[(size_t)g * OUT_N + nn];
        #pragma unroll
        for (int r = 0; r < 8; ++r) {
            int rr  = r + (hi ? 8 : 0);
            int row = s_rows[rr];
            if (row >= 0) out[(size_t)row * OUT_N + nn] = acc2[r] + bias;
        }
    }
}

// ---------------------------------------------------------------------------
// Launch
// ---------------------------------------------------------------------------
extern "C" void kernel_launch(void* const* d_in, const int* in_sizes, int n_in,
                              void* d_out, int out_size, void* d_ws, size_t ws_size,
                              hipStream_t stream) {
    const float* x   = (const float*)d_in[0];
    const int*   gid = (const int*)  d_in[1];
    const float* W1  = (const float*)d_in[2];
    const float* b1  = (const float*)d_in[3];
    const float* W2  = (const float*)d_in[4];
    const float* b2  = (const float*)d_in[5];
    float* out = (float*)d_out;
    int*   ws  = (int*)d_ws;

    route_init   <<<(WS_TOTAL + 255) / 256, 256, 0, stream>>>(ws);
    route_hist   <<<B_N / 256,             256, 0, stream>>>(gid, ws);
    route_plan   <<<1,                      64, 0, stream>>>(ws);
    route_scatter<<<B_N / 256,             256, 0, stream>>>(gid, ws);
    moe_mlp_kernel<<<MAX_TILES, 256, 0, stream>>>(ws, x, W1, b1, W2, b2, out);
}
